// LSTM_23424751632832
// MI455X (gfx1250) — compile-verified, matrix operand order checked by reading, not run on hardware
//
#include <hip/hip_runtime.h>
#include <hip/hip_bf16.h>

// MI455X / gfx1250:
//  - conv+bias+relu+double-maxpool fused as bf16 WMMA implicit GEMM, conv tile
//    staged in the 320KB WGP LDS (double pool == max over 13x13 conv patch)
//  - reduction FC (K=23328): split-K x27 WMMA GEMM + deterministic reduce
//    (no float atomics: atomic order would break run-to-run determinism)
//  - LSTM recurrence: persistent 1024-thread workgroup; recurrent GEMV runs on
//    WMMA with h replicated across B columns; Whh fragment-ordered bf16 in L2
//  - all WMMA paths: bf16 inputs, fp32 accumulate

typedef __bf16 bf16_t;
typedef __attribute__((ext_vector_type(16))) __bf16 v16bf;
typedef __attribute__((ext_vector_type(8)))  __bf16 v8bf;
typedef __attribute__((ext_vector_type(8)))  float  v8f;

#define S_SEQ 256
#define HIDN  256
#define G4H   1024
#define OUT_N 128
#define CFLAT 23328   // 32*27*27
#define KCHUNKS 27    // split-K factor for the 23328-deep GEMM (864 = 27*32 each)

__device__ __forceinline__ float sigmoidf_(float x) { return 1.0f / (1.0f + __expf(-x)); }

// ---- WMMA fragment helpers (16-bit A/B layout: lane<16 -> K {0..7,16..23},
//      lane>=16 -> +8; element e<8 = first run, e>=8 = second run) -------------
__device__ __forceinline__ v16bf frag_from_bf16(const bf16_t* row, int kb, int hi)
{
    v8bf lo = *reinterpret_cast<const v8bf*>(row + kb + 8 * hi);
    v8bf hv = *reinterpret_cast<const v8bf*>(row + kb + 16 + 8 * hi);
    v16bf f;
    #pragma unroll
    for (int e = 0; e < 8; ++e) { f[e] = lo[e]; f[e + 8] = hv[e]; }
    return f;
}

__device__ __forceinline__ v16bf frag_from_f32(const float* row, int kb, int hi)
{
    const float* p = row + kb;
    float4 f0 = *reinterpret_cast<const float4*>(p + 8 * hi);
    float4 f1 = *reinterpret_cast<const float4*>(p + 8 * hi + 4);
    float4 f2 = *reinterpret_cast<const float4*>(p + 16 + 8 * hi);
    float4 f3 = *reinterpret_cast<const float4*>(p + 16 + 8 * hi + 4);
    v16bf b;
    b[0]  = (bf16_t)f0.x; b[1]  = (bf16_t)f0.y; b[2]  = (bf16_t)f0.z; b[3]  = (bf16_t)f0.w;
    b[4]  = (bf16_t)f1.x; b[5]  = (bf16_t)f1.y; b[6]  = (bf16_t)f1.z; b[7]  = (bf16_t)f1.w;
    b[8]  = (bf16_t)f2.x; b[9]  = (bf16_t)f2.y; b[10] = (bf16_t)f2.z; b[11] = (bf16_t)f2.w;
    b[12] = (bf16_t)f3.x; b[13] = (bf16_t)f3.y; b[14] = (bf16_t)f3.z; b[15] = (bf16_t)f3.w;
    return b;
}

// ---------------------------------------------------------------------------
// Kernel 1: fused conv(3x3,3->32,pad1) + bias + relu + pool(3,2) + pool(6,4)
// One workgroup per (image, pooled row). Implicit GEMM M=13*224, N=32, K=27->32.
// ---------------------------------------------------------------------------
#define IN_ELEMS   (15 * 3 * 228)
#define CONV_ELEMS (2912 * 32)
#define CONV_SMEM_BYTES ((IN_ELEMS + CONV_ELEMS) * 2)

__global__ __launch_bounds__(864, 1)
void conv_pool_wmma(const float* __restrict__ x,      // (256,224,224,3) NHWC
                    const float* __restrict__ cw,     // (32,3,3,3) OIHW
                    const float* __restrict__ cb,     // (32,)
                    bf16_t* __restrict__ act)         // (256, 23328) bf16
{
    extern __shared__ bf16_t smem[];
    bf16_t* inb   = smem;              // [15][3][228], input col stored at +1
    bf16_t* convb = smem + IN_ELEMS;   // [2912][32]

    const int py   = blockIdx.x;
    const int img  = blockIdx.y;
    const int tid  = threadIdx.x;      // 864 threads = 27 waves
    const int lane = tid & 31;
    const int wv   = tid >> 5;
    const int nlo  = lane & 15;
    const int hi   = lane >> 4;

    for (int i = tid; i < 15 * 224; i += 864) {
        int lr  = i / 224, col = i - lr * 224;
        int grw = py * 8 - 1 + lr;
        float v0 = 0.f, v1 = 0.f, v2 = 0.f;
        if (grw >= 0 && grw < 224) {
            const float* p = x + (((size_t)img * 224 + grw) * 224 + col) * 3;
            v0 = p[0]; v1 = p[1]; v2 = p[2];
        }
        inb[(lr * 3 + 0) * 228 + col + 1] = (bf16_t)v0;
        inb[(lr * 3 + 1) * 228 + col + 1] = (bf16_t)v1;
        inb[(lr * 3 + 2) * 228 + col + 1] = (bf16_t)v2;
    }
    for (int i = tid; i < 15 * 3; i += 864) {
        inb[i * 228 + 0]   = (bf16_t)0.f;
        inb[i * 228 + 225] = (bf16_t)0.f;
        inb[i * 228 + 226] = (bf16_t)0.f;
        inb[i * 228 + 227] = (bf16_t)0.f;
    }

    v16bf bw0, bw1;                    // weight fragments, K padded 27 -> 32
    #pragma unroll
    for (int e = 0; e < 16; ++e) {
        int k = (e < 8) ? (8 * hi + e) : (16 + 8 * hi + (e - 8));
        float w0 = (k < 27) ? cw[nlo * 27 + k]        : 0.f;
        float w1 = (k < 27) ? cw[(nlo + 16) * 27 + k] : 0.f;
        bw0[e] = (bf16_t)w0;
        bw1[e] = (bf16_t)w1;
    }
    __syncthreads();

    for (int mt = wv; mt < 182; mt += 27) {
        const int m    = mt * 16 + nlo;
        const int r    = m / 224;
        const int col  = m - r * 224;
        const int basO = r * 684 + col;

        v16bf a;
        #pragma unroll
        for (int e = 0; e < 16; ++e) {
            int k = (e < 8) ? (8 * hi + e) : (16 + 8 * hi + (e - 8));
            bf16_t v = (bf16_t)0.f;
            if (k < 27) {
                int ic = k / 9, rem = k - ic * 9, ky = rem / 3, kx = rem - ky * 3;
                v = inb[basO + (ky * 3 + ic) * 228 + kx];
            }
            a[e] = v;
        }
        v8f z = {};
        v8f d0 = __builtin_amdgcn_wmma_f32_16x16x32_bf16(false, a, false, bw0, (short)0, z, false, false);
        v8f d1 = __builtin_amdgcn_wmma_f32_16x16x32_bf16(false, a, false, bw1, (short)0, z, false, false);

        const float b0 = cb[nlo], b1 = cb[nlo + 16];
        #pragma unroll
        for (int e = 0; e < 8; ++e) {
            int mm = mt * 16 + e + 8 * hi;
            convb[mm * 32 + nlo]      = (bf16_t)(d0[e] + b0);
            convb[mm * 32 + nlo + 16] = (bf16_t)(d1[e] + b1);
        }
    }
    __syncthreads();

    {   // double maxpool == max over 13x13 conv patch; init 0 == ReLU
        const int c  = tid & 31;
        const int px = tid >> 5;
        float best = 0.f;
        for (int r = 0; r < 13; ++r) {
            const bf16_t* rowp = convb + (r * 224 + 8 * px) * 32 + c;
            #pragma unroll
            for (int q = 0; q < 13; ++q)
                best = fmaxf(best, (float)rowp[q * 32]);
        }
        act[(size_t)img * CFLAT + c * 729 + py * 27 + px] = (bf16_t)best;
    }
}

// ---------------------------------------------------------------------------
// Kernel 2: generic bf16 WMMA GEMM (short-K layers), dual-accumulator ILP.
// D(MxN) = A(MxK,bf16) @ W(NxK,f32)^T + b1 + b2, optional relu, f32/bf16 out.
// ---------------------------------------------------------------------------
template<bool RELU, bool WF, bool WB>
__global__ __launch_bounds__(32)
void wmma_gemm(const bf16_t* __restrict__ A, int lda,
               const float*  __restrict__ W, int ldw,
               const float*  __restrict__ b1,
               const float*  __restrict__ b2,
               float*  __restrict__ outF,
               bf16_t* __restrict__ outB,
               int N, int K)
{
    const int lane = threadIdx.x & 31;
    const int nlo  = lane & 15, hi = lane >> 4;
    const int n0   = blockIdx.x * 16;
    const int m0   = blockIdx.y * 16;

    const bf16_t* arow = A + (size_t)(m0 + nlo) * lda;
    const float*  wrow = W + (size_t)(n0 + nlo) * ldw;

    v8f acc0 = {}, acc1 = {};
    int kb = 0;
    for (; kb + 64 <= K; kb += 64) {
        acc0 = __builtin_amdgcn_wmma_f32_16x16x32_bf16(false, frag_from_bf16(arow, kb, hi),
                                                       false, frag_from_f32(wrow, kb, hi),
                                                       (short)0, acc0, false, false);
        acc1 = __builtin_amdgcn_wmma_f32_16x16x32_bf16(false, frag_from_bf16(arow, kb + 32, hi),
                                                       false, frag_from_f32(wrow, kb + 32, hi),
                                                       (short)0, acc1, false, false);
    }
    for (; kb < K; kb += 32)
        acc0 = __builtin_amdgcn_wmma_f32_16x16x32_bf16(false, frag_from_bf16(arow, kb, hi),
                                                       false, frag_from_f32(wrow, kb, hi),
                                                       (short)0, acc0, false, false);
    v8f acc = acc0 + acc1;

    const int n = n0 + nlo;
    const float bias = (b1 ? b1[n] : 0.f) + (b2 ? b2[n] : 0.f);
    #pragma unroll
    for (int e = 0; e < 8; ++e) {
        int   m = m0 + e + 8 * hi;
        float v = acc[e] + bias;
        if (RELU) v = fmaxf(v, 0.f);
        if (WF) outF[(size_t)m * N + n] = v;
        if (WB) outB[(size_t)m * N + n] = (bf16_t)v;
    }
}

// ---------------------------------------------------------------------------
// Kernel 3: split-K WMMA GEMM for the K=23328 reduction layer. 27 chunks of
// 864. Each (tile, chunk) wave writes an fp32 partial; deterministic reduce
// kernel then applies bias+relu and emits bf16.
// ---------------------------------------------------------------------------
__global__ __launch_bounds__(32)
void wmma_gemm_splitk(const bf16_t* __restrict__ A, int lda,
                      const float*  __restrict__ W, int ldw,
                      float* __restrict__ part,      // (chunks, M, N)
                      int M, int N, int kPerChunk)
{
    const int lane = threadIdx.x & 31;
    const int nlo  = lane & 15, hi = lane >> 4;
    const int n0   = blockIdx.x * 16;
    const int m0   = blockIdx.y * 16;
    const int ch   = blockIdx.z;
    const int k0   = ch * kPerChunk, k1 = k0 + kPerChunk;

    const bf16_t* arow = A + (size_t)(m0 + nlo) * lda;
    const float*  wrow = W + (size_t)(n0 + nlo) * ldw;

    v8f acc0 = {}, acc1 = {};
    int kb = k0;
    for (; kb + 64 <= k1; kb += 64) {
        if (kb + 256 < k1) {
            __builtin_prefetch(arow + kb + 256, 0, 1);   // global_prefetch_b8
            __builtin_prefetch(wrow + kb + 256, 0, 1);
        }
        acc0 = __builtin_amdgcn_wmma_f32_16x16x32_bf16(false, frag_from_bf16(arow, kb, hi),
                                                       false, frag_from_f32(wrow, kb, hi),
                                                       (short)0, acc0, false, false);
        acc1 = __builtin_amdgcn_wmma_f32_16x16x32_bf16(false, frag_from_bf16(arow, kb + 32, hi),
                                                       false, frag_from_f32(wrow, kb + 32, hi),
                                                       (short)0, acc1, false, false);
    }
    for (; kb < k1; kb += 32)
        acc0 = __builtin_amdgcn_wmma_f32_16x16x32_bf16(false, frag_from_bf16(arow, kb, hi),
                                                       false, frag_from_f32(wrow, kb, hi),
                                                       (short)0, acc0, false, false);
    v8f acc = acc0 + acc1;

    float* pdst = part + (size_t)ch * M * N;
    const int n = n0 + nlo;
    #pragma unroll
    for (int e = 0; e < 8; ++e) {
        int m = m0 + e + 8 * hi;
        pdst[(size_t)m * N + n] = acc[e];
    }
}

__global__ void splitk_reduce(const float* __restrict__ part,
                              const float* __restrict__ bias,
                              bf16_t* __restrict__ outB,
                              int MN, int N, int chunks)
{
    int i = blockIdx.x * 256 + threadIdx.x;
    if (i >= MN) return;
    float s = 0.f;
    for (int c = 0; c < chunks; ++c) s += part[(size_t)c * MN + i];
    s += bias[i % N];
    outB[i] = (bf16_t)fmaxf(s, 0.f);
}

// ---------------------------------------------------------------------------
// Kernel 4: repack Whh (1024x256 f32) into fragment-ordered bf16 so the scan's
// A-fragments are coalesced b128 loads:
//   idx = mt*4096 + kt*512 + hi*256 + half*128 + nlo*8 + e
//   value = Whh[(mt*16+nlo)*256 + kt*32 + half*16 + hi*8 + e]
// ---------------------------------------------------------------------------
__global__ void repack_whh(const float* __restrict__ Whh, bf16_t* __restrict__ whhF)
{
    int t = blockIdx.x * 256 + threadIdx.x;       // 0 .. 262143
    int e    = t & 7;
    int nlo  = (t >> 3) & 15;
    int half = (t >> 7) & 1;
    int hi   = (t >> 8) & 1;
    int kt   = (t >> 9) & 7;
    int mt   = t >> 12;
    int row  = mt * 16 + nlo;
    int k    = kt * 32 + half * 16 + hi * 8 + e;
    whhF[t] = (bf16_t)Whh[row * 256 + k];
}

// ---------------------------------------------------------------------------
// Kernel 5: LSTM scan, recurrent GEMV on WMMA. 1024 threads = 32 waves, each
// wave owns 2 of the 64 gate M-tiles. B fragment = h (bf16, LDS) replicated
// across all 16 columns; lanes with nlo==0 extract column 0 of D.
// h/c state lives in registers of threads 0..255 for all 256 steps.
// ---------------------------------------------------------------------------
__global__ __launch_bounds__(1024, 1)
void lstm_scan_wmma(const float* __restrict__ xg,     // (256,1024)
                    const bf16_t* __restrict__ whhF,  // fragment-ordered
                    bf16_t* __restrict__ xnext,       // (256,256) bf16
                    float* __restrict__ h_out,        // (256,)
                    float* __restrict__ c_out)        // (256,)
{
    __shared__ __align__(16) bf16_t hbf[HIDN];
    __shared__ float gl[G4H];                        // recurrent gate part

    const int tid  = threadIdx.x;
    const int lane = tid & 31;
    const int wv   = tid >> 5;                       // 0..31
    const int nlo  = lane & 15, hi = lane >> 4;
    const int mt0  = 2 * wv, mt1 = 2 * wv + 1;

    const bf16_t* f0base = whhF + (size_t)mt0 * 4096 + hi * 256 + nlo * 8;
    const bf16_t* f1base = whhF + (size_t)mt1 * 4096 + hi * 256 + nlo * 8;

    float hreg = 0.f, creg = 0.f;
    if (tid < HIDN) hbf[tid] = (bf16_t)0.f;
    __syncthreads();

    for (int t = 0; t < S_SEQ; ++t) {
        v8f acc0 = {}, acc1 = {};
        #pragma unroll
        for (int kt = 0; kt < 8; ++kt) {
            // B fragment: h replicated across columns (same k-map as A)
            v8bf blo = *reinterpret_cast<const v8bf*>(&hbf[kt * 32 + 8 * hi]);
            v8bf bhi = *reinterpret_cast<const v8bf*>(&hbf[kt * 32 + 16 + 8 * hi]);
            v16bf b;
            #pragma unroll
            for (int e = 0; e < 8; ++e) { b[e] = blo[e]; b[e + 8] = bhi[e]; }

            const bf16_t* p0 = f0base + kt * 512;
            const bf16_t* p1 = f1base + kt * 512;
            v8bf a0l = *reinterpret_cast<const v8bf*>(p0);
            v8bf a0h = *reinterpret_cast<const v8bf*>(p0 + 128);
            v8bf a1l = *reinterpret_cast<const v8bf*>(p1);
            v8bf a1h = *reinterpret_cast<const v8bf*>(p1 + 128);
            v16bf a0, a1;
            #pragma unroll
            for (int e = 0; e < 8; ++e) {
                a0[e] = a0l[e]; a0[e + 8] = a0h[e];
                a1[e] = a1l[e]; a1[e + 8] = a1h[e];
            }
            acc0 = __builtin_amdgcn_wmma_f32_16x16x32_bf16(false, a0, false, b, (short)0, acc0, false, false);
            acc1 = __builtin_amdgcn_wmma_f32_16x16x32_bf16(false, a1, false, b, (short)0, acc1, false, false);
        }
        if (nlo == 0) {                 // lanes 0,16 hold column n=0
            #pragma unroll
            for (int e = 0; e < 8; ++e) {
                gl[mt0 * 16 + 8 * hi + e] = acc0[e];
                gl[mt1 * 16 + 8 * hi + e] = acc1[e];
            }
        }
        __syncthreads();
        if (tid < HIDN) {
            const float* xgt = xg + t * G4H;
            float ig = sigmoidf_(gl[tid]       + xgt[tid]);
            float fg = sigmoidf_(gl[tid + 256] + xgt[tid + 256]);
            float gg = tanhf    (gl[tid + 512] + xgt[tid + 512]);
            float og = sigmoidf_(gl[tid + 768] + xgt[tid + 768]);
            creg = fg * creg + ig * gg;
            hreg = og * tanhf(creg);
            bf16_t hb = (bf16_t)hreg;
            hbf[tid] = hb;
            xnext[t * HIDN + tid] = hb;
        }
        __syncthreads();
    }
    if (tid < HIDN) { h_out[tid] = hreg; c_out[tid] = creg; }
}

// ---------------------------------------------------------------------------
extern "C" void kernel_launch(void* const* d_in, const int* in_sizes, int n_in,
                              void* d_out, int out_size, void* d_ws, size_t ws_size,
                              hipStream_t stream)
{
    (void)in_sizes; (void)n_in; (void)out_size; (void)ws_size;

    const float* x      = (const float*)d_in[0];
    const float* conv_w = (const float*)d_in[1];
    const float* conv_b = (const float*)d_in[2];
    const float* red_w  = (const float*)d_in[3];
    const float* red_b  = (const float*)d_in[4];
    const float* fc_w   = (const float*)d_in[21];
    const float* fc_b   = (const float*)d_in[22];
    float* out = (float*)d_out;

    // workspace carve-out (~22 MB)
    char* ws = (char*)d_ws;
    bf16_t* act  = (bf16_t*)ws;  ws += (size_t)S_SEQ * CFLAT * 2;          // 11.94 MB
    bf16_t* xa   = (bf16_t*)ws;  ws += (size_t)S_SEQ * HIDN  * 2;
    bf16_t* xb   = (bf16_t*)ws;  ws += (size_t)S_SEQ * HIDN  * 2;
    float*  xg   = (float*)ws;   ws += (size_t)S_SEQ * G4H   * 4;          // 1 MB
    bf16_t* whhF = (bf16_t*)ws;  ws += (size_t)G4H * HIDN * 2;             // 0.5 MB
    float*  part = (float*)ws;   ws += (size_t)KCHUNKS * S_SEQ * HIDN * 4; // 6.75 MB

    (void)hipFuncSetAttribute(reinterpret_cast<const void*>(conv_pool_wmma),
                              hipFuncAttributeMaxDynamicSharedMemorySize,
                              CONV_SMEM_BYTES);

    // 1) fused conv + relu + double pool -> bf16 activations (256 x 23328)
    conv_pool_wmma<<<dim3(27, 256), 864, CONV_SMEM_BYTES, stream>>>(x, conv_w, conv_b, act);

    // 2) reduction FC via split-K (27 x 864) + deterministic reduce (+bias,relu)
    wmma_gemm_splitk<<<dim3(HIDN / 16, S_SEQ / 16, KCHUNKS), 32, 0, stream>>>(
        act, CFLAT, red_w, CFLAT, part, S_SEQ, HIDN, CFLAT / KCHUNKS);
    splitk_reduce<<<dim3((S_SEQ * HIDN + 255) / 256), 256, 0, stream>>>(
        part, red_b, xa, S_SEQ * HIDN, HIDN, KCHUNKS);

    // 3) 4-layer LSTM
    bf16_t* xcur = xa; bf16_t* xnxt = xb;
    for (int l = 0; l < 4; ++l) {
        const float* Wih = (const float*)d_in[5 + 4 * l + 0];
        const float* Whh = (const float*)d_in[5 + 4 * l + 1];
        const float* bih = (const float*)d_in[5 + 4 * l + 2];
        const float* bhh = (const float*)d_in[5 + 4 * l + 3];

        repack_whh<<<dim3(1024), 256, 0, stream>>>(Whh, whhF);
        wmma_gemm<false, true, false><<<dim3(G4H / 16, S_SEQ / 16), 32, 0, stream>>>(
            xcur, HIDN, Wih, HIDN, bih, bhh, xg, nullptr, G4H, HIDN);
        lstm_scan_wmma<<<dim3(1), 1024, 0, stream>>>(
            xg, whhF, xnxt,
            out + S_SEQ * OUT_N + l * HIDN,
            out + S_SEQ * OUT_N + 4 * HIDN + l * HIDN);

        bf16_t* t = xcur; xcur = xnxt; xnxt = t;
    }

    // 4) final FC: xcur @ fc_w^T + fc_b -> out (256 x 128, fp32)
    wmma_gemm<false, true, false><<<dim3(OUT_N / 16, S_SEQ / 16), 32, 0, stream>>>(
        xcur, HIDN, fc_w, HIDN, fc_b, nullptr, out, nullptr, OUT_N, HIDN);
}
